// MultiHeadAttention_14001593385413
// MI455X (gfx1250) — matrix operand order repaired and checked by reading, not compiled, not run on hardware
//
#include <hip/hip_runtime.h>

// ---------------------------------------------------------------------------
// MHA (causal) for MI455X gfx1250: bf16 WMMA + async global->LDS staging.
//   B=4, S=2048, D=1024, H=16, dh=64
// ---------------------------------------------------------------------------

#define BATCH     4
#define SEQ       2048
#define DMODEL    1024
#define NHEADS    16
#define HEADDIM   64
#define NEG_INF   (-1000000000.0f)

typedef __attribute__((ext_vector_type(8)))  __bf16 v8bf;
typedef __attribute__((ext_vector_type(16))) __bf16 v16bf;
typedef __attribute__((ext_vector_type(8)))  float  v8f;
typedef int v4i __attribute__((vector_size(16)));

union FragU { v16bf v; v8bf h[2]; };

// ---- CDNA5 async global->LDS copy (ASYNCcnt path), with sync fallback -----
#if defined(__has_builtin)
#if __has_builtin(__builtin_amdgcn_global_load_async_to_lds_b128)
#define HAVE_ASYNC_LDS 1
#endif
#endif
#ifndef HAVE_ASYNC_LDS
#define HAVE_ASYNC_LDS 0
#endif

#define AS1 __attribute__((address_space(1)))
#define AS3 __attribute__((address_space(3)))

__device__ __forceinline__ void async_cp16(void* lds, const void* g) {
#if HAVE_ASYNC_LDS
  // 16B per lane, memory -> LDS, no VGPR staging; tracked by ASYNCcnt
  __builtin_amdgcn_global_load_async_to_lds_b128(
      (AS1 v4i*)(uintptr_t)g, (AS3 v4i*)(uintptr_t)lds,
      /*imm offset=*/0, /*cpol=*/0);
#else
  *reinterpret_cast<v8bf*>(lds) = *reinterpret_cast<const v8bf*>(g);
#endif
}

__device__ __forceinline__ void async_wait0() {
#if HAVE_ASYNC_LDS
  asm volatile("s_wait_asynccnt 0" ::: "memory");
#endif
}

// ---------------------------------------------------------------------------
__device__ __forceinline__ __bf16 f2bf(float f) {
  unsigned u = __builtin_bit_cast(unsigned, f);
  u += 0x7fffu + ((u >> 16) & 1u);           // round-to-nearest-even
  unsigned short hs = (unsigned short)(u >> 16);
  return __builtin_bit_cast(__bf16, hs);
}

__device__ __forceinline__ v8f wmma_bf16(v16bf a, v16bf b, v8f c) {
  return __builtin_amdgcn_wmma_f32_16x16x32_bf16(
      false, a, false, b, (short)0, c, false, false);
}

// A-fragment (16x32 bf16) from row-major [rows][rowStride] tile.
__device__ __forceinline__ v16bf load_frag_A(const __bf16* tile, int rowStride,
                                             int row0, int kbase) {
  int lane = threadIdx.x & 31;
  int hi = lane >> 4, lr = lane & 15;
  const __bf16* p = tile + (size_t)(row0 + lr) * rowStride + kbase + hi * 8;
  FragU f;
  f.h[0] = *reinterpret_cast<const v8bf*>(p);
  f.h[1] = *reinterpret_cast<const v8bf*>(p + 16);
  return f.v;
}

// B-fragment (32x16 bf16) from an "n-major" tile [n][rowStride over k].
__device__ __forceinline__ v16bf load_frag_B(const __bf16* tile, int rowStride,
                                             int n0, int kbase) {
  int lane = threadIdx.x & 31;
  int hi = lane >> 4, lr = lane & 15;
  const __bf16* p = tile + (size_t)(n0 + lr) * rowStride + kbase + hi * 8;
  FragU f;
  f.h[0] = *reinterpret_cast<const v8bf*>(p);
  f.h[1] = *reinterpret_cast<const v8bf*>(p + 16);
  return f.v;
}

// ---------------------------------------------------------------------------
// f32 -> bf16 convert
// ---------------------------------------------------------------------------
__global__ __launch_bounds__(256) void cvt_bf16_kernel(const float* __restrict__ in,
                                                       __bf16* __restrict__ out, int n) {
  int i = blockIdx.x * blockDim.x + threadIdx.x;
  int stride = gridDim.x * blockDim.x;
  for (; i < n; i += stride) out[i] = f2bf(in[i]);
}

// ---------------------------------------------------------------------------
// GEMM: C[M x N] = A[M x K] * W[N x K]^T   (x @ W^T), K = N = 1024
// MODE 0: bf16 store into head-interleaved [b][h][s][dh] layout (Q/K/V)
// MODE 1: f32 row-major store (final projection)
// 128x128 block, 256 threads = 8 waves (2M x 4N), wave tile 64x32.
// K-chunk 64 (2 WMMA steps/tile), double-buffered async LDS staging.
// ---------------------------------------------------------------------------
template <int MODE>
__global__ __launch_bounds__(256) void gemm_xwT_kernel(const __bf16* __restrict__ A,
                                                       const __bf16* __restrict__ W,
                                                       void* __restrict__ Cout) {
  __shared__ __align__(16) __bf16 As[2][128 * 64];
  __shared__ __align__(16) __bf16 Bs[2][128 * 64];

  const int tid = threadIdx.x;
  const int m0 = blockIdx.y * 128;
  const int n0 = blockIdx.x * 128;
  const int w = tid >> 5, lane = tid & 31;
  const int waveM = w & 1;     // 64 rows each
  const int waveN = w >> 1;    // 32 cols each
  const int hi = lane >> 4, lr = lane & 15;

  // per-thread staging addresses, computed once (16B chunk each, x4)
  const __bf16* pA[4];
  const __bf16* pB[4];
  int ldsOff[4];
#pragma unroll
  for (int l = 0; l < 4; ++l) {
    int lin = tid + l * 256;          // 1024 chunks of 8 bf16 = 128x64 tile
    int row = lin >> 3, seg = lin & 7;
    ldsOff[l] = row * 64 + seg * 8;
    pA[l] = A + (size_t)(m0 + row) * DMODEL + seg * 8;
    pB[l] = W + (size_t)(n0 + row) * DMODEL + seg * 8;
  }

  v8f c[4][2] = {};

  const int KC = 64;
  const int NITER = DMODEL / KC;      // 16

  // prologue: stage tile 0
#pragma unroll
  for (int l = 0; l < 4; ++l) {
    async_cp16(&As[0][ldsOff[l]], pA[l]);
    async_cp16(&Bs[0][ldsOff[l]], pB[l]);
  }

  for (int it = 0; it < NITER; ++it) {
    async_wait0();
    __syncthreads();                  // tile `it` ready; buffer (it+1)&1 free

    if (it + 1 < NITER) {
      const int k0 = (it + 1) * KC;
      const int nb = (it + 1) & 1;
#pragma unroll
      for (int l = 0; l < 4; ++l) {
        async_cp16(&As[nb][ldsOff[l]], pA[l] + k0);
        async_cp16(&Bs[nb][ldsOff[l]], pB[l] + k0);
      }
    }

    const int cur = it & 1;
#pragma unroll
    for (int kk = 0; kk < 2; ++kk) {
      v16bf bfrag[2];
#pragma unroll
      for (int nt = 0; nt < 2; ++nt)
        bfrag[nt] = load_frag_B(&Bs[cur][0], 64, waveN * 32 + nt * 16, kk * 32);
#pragma unroll
      for (int mt = 0; mt < 4; ++mt) {
        v16bf afrag = load_frag_A(&As[cur][0], 64, waveM * 64 + mt * 16, kk * 32);
#pragma unroll
        for (int nt = 0; nt < 2; ++nt)
          c[mt][nt] = wmma_bf16(afrag, bfrag[nt], c[mt][nt]);
      }
    }
  }

  // store: C layout -> VGPR r, lanes0-15: (M=r, N=lr); lanes16-31: (M=r+8, N=lr)
#pragma unroll
  for (int mt = 0; mt < 4; ++mt) {
#pragma unroll
    for (int nt = 0; nt < 2; ++nt) {
#pragma unroll
      for (int r = 0; r < 8; ++r) {
        int gm = m0 + waveM * 64 + mt * 16 + r + 8 * hi;
        int ge = n0 + waveN * 32 + nt * 16 + lr;
        float val = c[mt][nt][r];
        if (MODE == 0) {
          int b = gm >> 11, s = gm & (SEQ - 1);
          int hh = ge >> 6, dh = ge & 63;
          ((__bf16*)Cout)[((((size_t)b * NHEADS + hh) * SEQ) + s) * HEADDIM + dh] = f2bf(val);
        } else {
          ((float*)Cout)[(size_t)gm * DMODEL + ge] = val;
        }
      }
    }
  }
}

// ---------------------------------------------------------------------------
// Causal flash attention.
// Grid: x = SEQ/64 query blocks, y = BATCH*NHEADS. Block = 128 threads (4 waves).
// Each wave owns 16 query rows; key blocks of 64 staged in LDS
// (K tile async-copied; V tile transposed through VGPRs).
// ---------------------------------------------------------------------------
__global__ __launch_bounds__(128) void attn_kernel(const __bf16* __restrict__ Q,
                                                   const __bf16* __restrict__ K,
                                                   const __bf16* __restrict__ V,
                                                   __bf16* __restrict__ O) {
  __shared__ __align__(16) __bf16 Ks[64 * 64];          // [key][d]
  __shared__ __align__(16) __bf16 Vt[64 * 64];          // [d][key]
  __shared__ __align__(16) float  Sst[4][16 * 64];
  __shared__ __align__(16) __bf16 Pst[4][16 * 64];
  __shared__ float Alp[4][16];

  const int tid = threadIdx.x;
  const int w = tid >> 5, lane = tid & 31;
  const int hi = lane >> 4, lr = lane & 15;
  const int bh = blockIdx.y;
  const int b = bh >> 4, h = bh & 15;
  const int qrow0 = blockIdx.x * 64 + w * 16;
  const size_t baseBH = (size_t)bh * SEQ * HEADDIM;

  // per-thread K/V staging addresses (advance by 64 rows per key block)
  int kkey[4], kseg[4];
#pragma unroll
  for (int i = 0; i < 4; ++i) {
    int lin = tid + i * 128;
    kkey[i] = lin >> 3;
    kseg[i] = lin & 7;
  }

  // Q fragments (16 rows x 64 d) -> two 16x32 A-fragments
  v16bf qf[2];
#pragma unroll
  for (int kc = 0; kc < 2; ++kc) {
    const __bf16* qp = Q + baseBH + (size_t)(qrow0 + lr) * HEADDIM + kc * 32 + hi * 8;
    FragU f;
    f.h[0] = *reinterpret_cast<const v8bf*>(qp);
    f.h[1] = *reinterpret_cast<const v8bf*>(qp + 16);
    qf[kc] = f.v;
  }

  v8f o[4] = {};
  float mrow = -3.0e38f, lrow = 0.0f;   // meaningful in lanes 0..15

  const int nblocks = blockIdx.x + 1;   // causal
  for (int jb = 0; jb < nblocks; ++jb) {
    const int kb0 = jb * 64;
    __syncthreads();                    // previous iteration done with Ks/Vt

    // K tile: async global->LDS; V tile: transpose through VGPRs
#pragma unroll
    for (int i = 0; i < 4; ++i) {
      const size_t gidx = baseBH + (size_t)(kb0 + kkey[i]) * HEADDIM + kseg[i] * 8;
      async_cp16(&Ks[kkey[i] * 64 + kseg[i] * 8], &K[gidx]);
      v8bf vv = *reinterpret_cast<const v8bf*>(&V[gidx]);
#pragma unroll
      for (int u = 0; u < 8; ++u) Vt[(kseg[i] * 8 + u) * 64 + kkey[i]] = vv[u];
    }
    async_wait0();
    __syncthreads();

    // scores: S(16x64) = Q x K^T
#pragma unroll
    for (int nt = 0; nt < 4; ++nt) {
      v8f s = {};
#pragma unroll
      for (int kc = 0; kc < 2; ++kc) {
        v16bf bfrag = load_frag_B(Ks, 64, nt * 16, kc * 32);
        s = wmma_bf16(qf[kc], bfrag, s);
      }
#pragma unroll
      for (int r = 0; r < 8; ++r)
        Sst[w][(r + 8 * hi) * 64 + nt * 16 + lr] = s[r];
    }
    __syncthreads();

    // online softmax: lanes 0..15 each own one query row
    if (hi == 0) {
      const int grow = qrow0 + lr;
      float* sp = &Sst[w][lr * 64];
      float mx = mrow;
#pragma unroll 8
      for (int j = 0; j < 64; ++j) {
        float x = sp[j] * 0.125f;                 // 1/sqrt(64)
        if (kb0 + j > grow) x = NEG_INF;
        sp[j] = x;
        mx = fmaxf(mx, x);
      }
      float a = __expf(mrow - mx);
      float sum = 0.0f;
#pragma unroll 8
      for (int j = 0; j < 64; ++j) {
        float e = __expf(sp[j] - mx);
        sum += e;
        Pst[w][lr * 64 + j] = f2bf(e);
      }
      lrow = lrow * a + sum;
      mrow = mx;
      Alp[w][lr] = a;
    }
    __syncthreads();

    // rescale O, then O += P(16x64) x V(64x64)
    v16bf pf[2];
#pragma unroll
    for (int kc = 0; kc < 2; ++kc) pf[kc] = load_frag_A(Pst[w], 64, 0, kc * 32);
#pragma unroll
    for (int nt = 0; nt < 4; ++nt) {
      v8f acc = o[nt];
#pragma unroll
      for (int r = 0; r < 8; ++r) acc[r] *= Alp[w][r + 8 * hi];
#pragma unroll
      for (int kc = 0; kc < 2; ++kc) {
        v16bf vfrag = load_frag_B(Vt, 64, nt * 16, kc * 32);
        acc = wmma_bf16(pf[kc], vfrag, acc);
      }
      o[nt] = acc;
    }
  }

  // finalize
  if (hi == 0) Alp[w][lr] = 1.0f / lrow;
  __syncthreads();
#pragma unroll
  for (int nt = 0; nt < 4; ++nt) {
#pragma unroll
    for (int r = 0; r < 8; ++r) {
      float val = o[nt][r] * Alp[w][r + 8 * hi];
      int s = qrow0 + r + 8 * hi;
      int dh = nt * 16 + lr;
      O[((size_t)(b * SEQ + s)) * DMODEL + h * HEADDIM + dh] = f2bf(val);
    }
  }
}

// ---------------------------------------------------------------------------
// Launcher
// ---------------------------------------------------------------------------
extern "C" void kernel_launch(void* const* d_in, const int* in_sizes, int n_in,
                              void* d_out, int out_size, void* d_ws, size_t ws_size,
                              hipStream_t stream) {
  (void)in_sizes; (void)n_in; (void)out_size; (void)ws_size;

  const float* x  = (const float*)d_in[0];
  const float* wq = (const float*)d_in[1];
  const float* wk = (const float*)d_in[2];
  const float* wv = (const float*)d_in[3];
  const float* wo = (const float*)d_in[4];
  float* out = (float*)d_out;

  const size_t XN = (size_t)BATCH * SEQ * DMODEL;
  const size_t WN = (size_t)DMODEL * DMODEL;

  char* ws = (char*)d_ws;
  __bf16* xb  = (__bf16*)(ws);                       size_t off = XN * 2;
  __bf16* wqb = (__bf16*)(ws + off);                 off += WN * 2;
  __bf16* wkb = (__bf16*)(ws + off);                 off += WN * 2;
  __bf16* wvb = (__bf16*)(ws + off);                 off += WN * 2;
  __bf16* wob = (__bf16*)(ws + off);                 off += WN * 2;
  __bf16* Qb  = (__bf16*)(ws + off);                 off += XN * 2;
  __bf16* Kb  = (__bf16*)(ws + off);                 off += XN * 2;
  __bf16* Vb  = (__bf16*)(ws + off);                 off += XN * 2;
  __bf16* Ob  = (__bf16*)(ws + off);                 off += XN * 2;

  {
    int blk = 256;
    hipLaunchKernelGGL(cvt_bf16_kernel, dim3((XN + 255) / 256), dim3(blk), 0, stream, x,  xb,  (int)XN);
    hipLaunchKernelGGL(cvt_bf16_kernel, dim3((WN + 255) / 256), dim3(blk), 0, stream, wq, wqb, (int)WN);
    hipLaunchKernelGGL(cvt_bf16_kernel, dim3((WN + 255) / 256), dim3(blk), 0, stream, wk, wkb, (int)WN);
    hipLaunchKernelGGL(cvt_bf16_kernel, dim3((WN + 255) / 256), dim3(blk), 0, stream, wv, wvb, (int)WN);
    hipLaunchKernelGGL(cvt_bf16_kernel, dim3((WN + 255) / 256), dim3(blk), 0, stream, wo, wob, (int)WN);
  }

  {
    dim3 g(DMODEL / 128, (BATCH * SEQ) / 128);   // (8, 64)
    dim3 blk(256);
    hipLaunchKernelGGL((gemm_xwT_kernel<0>), g, blk, 0, stream, xb, wqb, (void*)Qb);
    hipLaunchKernelGGL((gemm_xwT_kernel<0>), g, blk, 0, stream, xb, wkb, (void*)Kb);
    hipLaunchKernelGGL((gemm_xwT_kernel<0>), g, blk, 0, stream, xb, wvb, (void*)Vb);
  }

  {
    dim3 g(SEQ / 64, BATCH * NHEADS);            // (32, 64)
    hipLaunchKernelGGL(attn_kernel, g, dim3(128), 0, stream, Qb, Kb, Vb, Ob);
  }

  {
    dim3 g(DMODEL / 128, (BATCH * SEQ) / 128);
    hipLaunchKernelGGL((gemm_xwT_kernel<1>), g, dim3(256), 0, stream, Ob, wob, (void*)out);
  }
}
